// MultiHeadSelfAttention_25821343383810
// MI455X (gfx1250) — matrix operand order repaired and checked by reading, not compiled, not run on hardware
//
#include <hip/hip_runtime.h>
#include <hip/hip_bf16.h>

#define NHEADS   16
#define D_MODEL  2048
#define Q_LORA   1536
#define KV_LORA  512
#define Q_HEAD   192
#define KV_HEAD  256
#define V_DIM    128
#define BATCH    2
#define SEQ      2048
#define BT       (BATCH * SEQ)          // 4096 rows
#define QN       (NHEADS * Q_HEAD)      // 3072
#define KVN      (NHEADS * KV_HEAD)     // 4096
#define OUT_N    (NHEADS * V_DIM)       // 2048

typedef __bf16 bf16;
typedef __attribute__((ext_vector_type(16))) __bf16 v16bf;
typedef __attribute__((ext_vector_type(8)))  __bf16 v8bf;
typedef __attribute__((ext_vector_type(8)))  float  v8f;
typedef unsigned int u32x4 __attribute__((ext_vector_type(4)));
typedef int          i32x8 __attribute__((ext_vector_type(8)));
typedef int          i32x4 __attribute__((ext_vector_type(4)));

__device__ inline v8f wmma_bf16(v16bf a, v16bf b, v8f c) {
  // D = A(16x32 bf16) * B(32x16 bf16) + C(16x16 f32)
  return __builtin_amdgcn_wmma_f32_16x16x32_bf16(false, a, false, b, (short)0, c,
                                                 false, false);
}

__device__ inline v16bf cat8(v8bf lo, v8bf hi) {
  v16bf r;
#pragma unroll
  for (int i = 0; i < 8; ++i) { r[i] = lo[i]; r[i + 8] = hi[i]; }
  return r;
}

// A fragment 16x32 (MxK) from row-major A, leading dim lda.
// lane<16: row=lane, elems 0-7 = K0-7, 8-15 = K16-23
// lane>=16: row=lane-16, elems 0-7 = K8-15, 8-15 = K24-31
__device__ inline v16bf load_a_frag(const bf16* A, int lda, int m0, int k0, int lane) {
  int m  = m0 + (lane & 15);
  int kb = k0 + ((lane & 16) ? 8 : 0);
  const bf16* p = A + (size_t)m * lda + kb;
  return cat8(*(const v8bf*)(p), *(const v8bf*)(p + 16));
}

// B fragment 32x16 (KxN) from N-major (transposed) Bt[N][K], leading dim ldb.
// lane<16: col=lane, elems 0-7 = K0-7, 8-15 = K16-23
// lane>=16: col=lane-16, elems 0-7 = K8-15, 8-15 = K24-31
__device__ inline v16bf load_b_frag(const bf16* Bt, int ldb, int n0, int k0, int lane) {
  int n  = n0 + (lane & 15);
  int kb = k0 + ((lane & 16) ? 8 : 0);
  const bf16* p = Bt + (size_t)n * ldb + kb;
  return cat8(*(const v8bf*)(p), *(const v8bf*)(p + 16));
}

// ---------------------------------------------------------------------------
// TDM: DMA a 64-row x 32-col bf16 tile of Bt[N][K] (starting at n0,k0) to LDS.
// D# per CDNA5 ISA sect. 8 (2D tensor, data_size=2B).
// clang-23 amdgpu-toolchain form: 6 args (g0, g1, g2, g3, extra, cpol).
// ---------------------------------------------------------------------------
__device__ inline void tdm_load_btile(const bf16* Bt, int N, int K,
                                      int n0, int k0, unsigned lds_off) {
  unsigned long long ga = (unsigned long long)(size_t)(Bt + (size_t)n0 * K + k0);
  u32x4 g0;
  g0[0] = 1u;                                   // count=1 (valid user D#)
  g0[1] = lds_off;                              // lds_addr (bytes)
  g0[2] = (unsigned)ga;                         // global_addr[31:0]
  g0[3] = (unsigned)((ga >> 32) & 0x1FFFFFFull) // global_addr[56:32]
          | (2u << 30);                         // type = 2 ("image")
  i32x8 g1;
  g1[0] = 0x10000;                              // data_size=1 -> 2 bytes
  g1[1] = (int)((unsigned)(K & 0xFFFF) << 16);  // tensor_dim0[15:0]
  g1[2] = (int)(((unsigned)K >> 16) | ((unsigned)(N & 0xFFFF) << 16)); // dim0 hi | dim1 lo
  g1[3] = (int)(((unsigned)N >> 16) | (32u << 16)); // dim1 hi | tile_dim0=32
  g1[4] = 64;                                   // tile_dim1=64 (tile_dim2=0 -> 2D)
  g1[5] = K;                                    // tensor_dim0_stride (elements)
  g1[6] = 0;
  g1[7] = 0;
  i32x4 z4 = {0, 0, 0, 0};
  i32x8 z8 = {0, 0, 0, 0, 0, 0, 0, 0};
  __builtin_amdgcn_tensor_load_to_lds(g0, g1, z4, z4, z8, 0);
}

// ---------------------------------------------------------------------------
// Generic bf16 GEMM: C[M,N] f32 = A[M,K] bf16 (row-major) * Bt[N,K] bf16.
// Block: 256 threads = 8 waves, each wave computes 16x64 -> block tile 128x64.
// B tile (64x32) is staged to LDS by the Tensor Data Mover, double-buffered:
// wave 0 issues tensor_load_to_lds for step k+1 while all waves run WMMAs on
// step k from LDS; completion via s_wait_tensorcnt + workgroup barrier.
// Requires M%128==0, N%64==0, K%32==0 (true for every call here).
// ---------------------------------------------------------------------------
__global__ __launch_bounds__(256)
void gemm_bf16_nt(const bf16* __restrict__ A, const bf16* __restrict__ Bt,
                  float* __restrict__ C, int M, int N, int K) {
  __shared__ __align__(16) bf16 btile[2][64][32];
  int lane = threadIdx.x & 31;
  int wv   = threadIdx.x >> 5;
  int m0   = blockIdx.y * 128 + wv * 16;
  int n0   = blockIdx.x * 64;
  bool issuer = (wv == 0);

  if (issuer) {
    unsigned loff0 = (unsigned)(size_t)(void*)&btile[0][0][0];
    tdm_load_btile(Bt, N, K, n0, 0, loff0);
    __builtin_amdgcn_s_wait_tensorcnt(0);
  }
  __syncthreads();

  v8f c0 = {0.f,0.f,0.f,0.f,0.f,0.f,0.f,0.f};
  v8f c1 = c0, c2 = c0, c3 = c0;
  for (int k0 = 0; k0 < K; k0 += 32) {
    int cur = (k0 >> 5) & 1;
    if (issuer && (k0 + 32 < K)) {
      unsigned loffn = (unsigned)(size_t)(void*)&btile[cur ^ 1][0][0];
      tdm_load_btile(Bt, N, K, n0, k0 + 32, loffn);
    }
    if (k0 + 64 < K)   // prefetch A two steps ahead (global_prefetch_b8)
      __builtin_prefetch(A + (size_t)(m0 + (lane & 15)) * K + k0 + 64, 0, 0);

    const bf16* bl = &btile[cur][0][0];
    v16bf a = load_a_frag(A, K, m0, k0, lane);
    c0 = wmma_bf16(a, load_b_frag(bl, 32,  0, 0, lane), c0);
    c1 = wmma_bf16(a, load_b_frag(bl, 32, 16, 0, lane), c1);
    c2 = wmma_bf16(a, load_b_frag(bl, 32, 32, 0, lane), c2);
    c3 = wmma_bf16(a, load_b_frag(bl, 32, 48, 0, lane), c3);

    if (issuer) __builtin_amdgcn_s_wait_tensorcnt(0);
    __syncthreads();
  }

  int nc = n0 + (lane & 15);
  int mb = m0 + ((lane >> 4) & 1) * 8;
#pragma unroll
  for (int r = 0; r < 8; ++r) {
    float* crow = C + (size_t)(mb + r) * N + nc;
    crow[0] = c0[r]; crow[16] = c1[r]; crow[32] = c2[r]; crow[48] = c3[r];
  }
}

// ---------------------------------------------------------------------------
// Elementwise helpers
// ---------------------------------------------------------------------------
__global__ void convert_bf16_kernel(const float* __restrict__ in,
                                    bf16* __restrict__ out, size_t n) {
  size_t i = (size_t)blockIdx.x * blockDim.x + threadIdx.x;
  if (i < n) out[i] = (bf16)in[i];
}

// in[K][N] f32  ->  out[N][K] bf16
__global__ void transpose_bf16_kernel(const float* __restrict__ in,
                                      bf16* __restrict__ out, int K, int N) {
  size_t i = (size_t)blockIdx.x * blockDim.x + threadIdx.x;
  if (i >= (size_t)K * N) return;
  int k = (int)(i / N), n = (int)(i % N);
  out[(size_t)n * K + k] = (bf16)in[i];
}

// Row-wise RMSNorm: out[row][0..len) bf16 = norm(in[row*in_stride + 0..len)) * w
__global__ __launch_bounds__(256)
void rmsnorm_bf16_kernel(const float* __restrict__ in, const float* __restrict__ w,
                         bf16* __restrict__ out, int len, int in_stride) {
  int row  = blockIdx.x;
  int lane = threadIdx.x & 31, wv = threadIdx.x >> 5;
  const float* x = in + (size_t)row * in_stride;
  __shared__ float red[8];
  float ss = 0.f;
  for (int i = threadIdx.x; i < len; i += 256) { float v = x[i]; ss += v * v; }
#pragma unroll
  for (int off = 16; off > 0; off >>= 1) ss += __shfl_xor(ss, off, 32);
  if (lane == 0) red[wv] = ss;
  __syncthreads();
  float tot = 0.f;
#pragma unroll
  for (int i = 0; i < 8; ++i) tot += red[i];
  float rinv = rsqrtf(tot / (float)len + 1e-6f);
  for (int i = threadIdx.x; i < len; i += 256)
    out[(size_t)row * len + i] = (bf16)(x[i] * rinv * w[i]);
}

__device__ inline float rope_val(const float* p, int rd, float pos) {
  int j = rd & 31;
  float inv = exp2f(-(float)j * 0.41524101186f);   // 10000^(-2j/64)
  float ang = pos * inv;
  float c = cosf(ang), s = sinf(ang);
  float x1 = p[j], x2 = p[32 + j];
  return (rd < 32) ? (x1 * c - x2 * s) : (x2 * c + x1 * s);
}

// q f32 [BT][3072] -> qs bf16 [B,H,T,192] with RoPE on dims 128..191
__global__ __launch_bounds__(256)
void build_q_kernel(const float* __restrict__ q, bf16* __restrict__ qs) {
  int bt = blockIdx.x;
  int b = bt / SEQ, t = bt % SEQ;
  const float* row = q + (size_t)bt * QN;
  for (int i = threadIdx.x; i < QN; i += 256) {
    int h = i / Q_HEAD, d = i % Q_HEAD;
    float val = (d < 128) ? row[i]
                          : rope_val(row + h * Q_HEAD + 128, d - 128, (float)t);
    qs[(((size_t)b * NHEADS + h) * SEQ + t) * Q_HEAD + d] = (bf16)val;
  }
}

// kvb f32 [BT][4096], kv f32 [BT][576] ->
//   ks bf16 [B,H,T,192] (nope 0..127 per-head, roped shared k_rope 128..191)
//   vT bf16 [B,H,128,T]
__global__ __launch_bounds__(256)
void build_kv_kernel(const float* __restrict__ kvb, const float* __restrict__ kv,
                     bf16* __restrict__ ks, bf16* __restrict__ vT) {
  int bt = blockIdx.x;
  int b = bt / SEQ, t = bt % SEQ;
  const float* rb = kvb + (size_t)bt * KVN;
  const float* rr = kv + (size_t)bt * (KV_LORA + 64) + KV_LORA;
  for (int i = threadIdx.x; i < NHEADS * KV_HEAD; i += 256) {
    int h = i >> 8, c = i & 255;
    float v = rb[i];
    if (c < 128)
      ks[(((size_t)b * NHEADS + h) * SEQ + t) * Q_HEAD + c] = (bf16)v;
    else
      vT[(((size_t)b * NHEADS + h) * V_DIM + (c - 128)) * SEQ + t] = (bf16)v;
  }
  for (int d = threadIdx.x; d < 64; d += 256) {
    bf16 v = (bf16)rope_val(rr, d, (float)t);
#pragma unroll
    for (int h = 0; h < NHEADS; ++h)
      ks[(((size_t)b * NHEADS + h) * SEQ + t) * Q_HEAD + 128 + d] = v;
  }
}

// ---------------------------------------------------------------------------
// Flash-style causal attention, one wave = 16 q rows, block = 4 waves = 64 rows.
// grid = (T/64, B*H). qs/ks: [B,H,T,192], vT: [B,H,128,T].
// Output: attn bf16 [B*T][H*128] (row-major, ready as GEMM A operand).
// ---------------------------------------------------------------------------
__global__ __launch_bounds__(128)
void mla_attn_kernel(const bf16* __restrict__ qs, const bf16* __restrict__ ks,
                     const bf16* __restrict__ vT, bf16* __restrict__ attn) {
  const float scale = 0.07216878365f;   // 1/sqrt(192)
  int lane = threadIdx.x & 31;
  int wv   = threadIdx.x >> 5;
  int bh   = blockIdx.y;
  int b    = bh / NHEADS, h = bh % NHEADS;
  int m0   = blockIdx.x * 64 + wv * 16;              // q-row base within T

  const bf16* qp = qs + (size_t)bh * SEQ * Q_HEAD;
  const bf16* kp = ks + (size_t)bh * SEQ * Q_HEAD;
  const bf16* vp = vT + (size_t)bh * V_DIM * SEQ;

  __shared__ __align__(16) bf16 plds[4][16][32];

  v16bf qf[6];
#pragma unroll
  for (int i = 0; i < 6; ++i) qf[i] = load_a_frag(qp, Q_HEAD, m0, i * 32, lane);

  v8f zero = {0.f,0.f,0.f,0.f,0.f,0.f,0.f,0.f};
  v8f o[8];
#pragma unroll
  for (int t = 0; t < 8; ++t) o[t] = zero;
  float mrow[8], lrow[8];
#pragma unroll
  for (int r = 0; r < 8; ++r) { mrow[r] = -1e30f; lrow[r] = 0.f; }

  int coll = lane & 15;
  int rowb = m0 + ((lane >> 4) & 1) * 8;   // global row of accumulator elem r

  for (int k0 = 0; k0 < m0 + 16; k0 += 32) {
    // ---- S = Q K^T over 32 key positions (two 16-wide tiles) ----
    v8f s0 = zero, s1 = zero;
#pragma unroll
    for (int i = 0; i < 6; ++i) {
      s0 = wmma_bf16(qf[i], load_b_frag(kp, Q_HEAD, k0,      i * 32, lane), s0);
      s1 = wmma_bf16(qf[i], load_b_frag(kp, Q_HEAD, k0 + 16, i * 32, lane), s1);
    }
    // ---- online softmax update ----
#pragma unroll
    for (int r = 0; r < 8; ++r) {
      int row = rowb + r;
      float v0 = s0[r] * scale;
      float v1 = s1[r] * scale;
      if (k0 + coll      > row) v0 = -1e30f;
      if (k0 + 16 + coll > row) v1 = -1e30f;
      float mx = fmaxf(v0, v1);
      mx = fmaxf(mx, __shfl_xor(mx, 1, 32));
      mx = fmaxf(mx, __shfl_xor(mx, 2, 32));
      mx = fmaxf(mx, __shfl_xor(mx, 4, 32));
      mx = fmaxf(mx, __shfl_xor(mx, 8, 32));
      float mnew  = fmaxf(mrow[r], mx);
      float alpha = __expf(mrow[r] - mnew);
      mrow[r] = mnew;
      float p0 = __expf(v0 - mnew);
      float p1 = __expf(v1 - mnew);
      float rs = p0 + p1;
      rs += __shfl_xor(rs, 1, 32);
      rs += __shfl_xor(rs, 2, 32);
      rs += __shfl_xor(rs, 4, 32);
      rs += __shfl_xor(rs, 8, 32);
      lrow[r] = lrow[r] * alpha + rs;
#pragma unroll
      for (int t = 0; t < 8; ++t) o[t][r] *= alpha;
      int rloc = r + ((lane >> 4) & 1) * 8;
      plds[wv][rloc][coll]      = (bf16)p0;     // transpose C-layout -> [row][col]
      plds[wv][rloc][coll + 16] = (bf16)p1;
    }
    // ---- O += P V : P is A-operand (16x32), V^T supplies B fragments ----
    v16bf pf = load_a_frag(&plds[wv][0][0], 32, 0, 0, lane);
#pragma unroll
    for (int t = 0; t < 8; ++t)
      o[t] = wmma_bf16(pf, load_b_frag(vp, SEQ, t * 16, k0, lane), o[t]);
  }

  // ---- normalize & store: attn[(b*T + row)][h*128 + d] ----
#pragma unroll
  for (int r = 0; r < 8; ++r) {
    float inv = 1.f / lrow[r];
    int row = rowb + r;
    bf16* arow = attn + ((size_t)b * SEQ + row) * OUT_N + h * V_DIM;
#pragma unroll
    for (int t = 0; t < 8; ++t)
      arow[t * 16 + coll] = (bf16)(o[t][r] * inv);
  }
}

// ---------------------------------------------------------------------------
extern "C" void kernel_launch(void* const* d_in, const int* in_sizes, int n_in,
                              void* d_out, int out_size, void* d_ws, size_t ws_size,
                              hipStream_t stream) {
  const float* x        = (const float*)d_in[0];
  const float* q_a_w    = (const float*)d_in[1];
  const float* q_a_nw   = (const float*)d_in[2];
  const float* q_b_w    = (const float*)d_in[3];
  const float* kv_a_w   = (const float*)d_in[4];
  const float* kv_nw    = (const float*)d_in[5];
  const float* kv_b_w   = (const float*)d_in[6];
  const float* o_w      = (const float*)d_in[7];
  float* out = (float*)d_out;

  char* base = (char*)d_ws;
  size_t off = 0;
  auto alloc = [&](size_t bytes) -> void* {
    void* p = base + off;
    off += (bytes + 255) & ~(size_t)255;
    return p;
  };

  bf16*  xb     = (bf16*)alloc((size_t)BT * D_MODEL * 2);
  bf16*  wqa_t  = (bf16*)alloc((size_t)D_MODEL * Q_LORA * 2);
  bf16*  wqb_t  = (bf16*)alloc((size_t)Q_LORA * QN * 2);
  bf16*  wkva_t = (bf16*)alloc((size_t)D_MODEL * (KV_LORA + 64) * 2);
  bf16*  wkvb_t = (bf16*)alloc((size_t)KV_LORA * KVN * 2);
  bf16*  wo_t   = (bf16*)alloc((size_t)OUT_N * D_MODEL * 2);
  float* qa_f   = (float*)alloc((size_t)BT * Q_LORA * 4);
  bf16*  qa_n   = (bf16*)alloc((size_t)BT * Q_LORA * 2);
  float* q_f    = (float*)alloc((size_t)BT * QN * 4);
  float* kv_f   = (float*)alloc((size_t)BT * (KV_LORA + 64) * 4);
  bf16*  kv_n   = (bf16*)alloc((size_t)BT * KV_LORA * 2);
  float* kvb_f  = (float*)alloc((size_t)BT * KVN * 4);
  bf16*  qsb    = (bf16*)alloc((size_t)BT * QN * 2);
  bf16*  ksb    = (bf16*)alloc((size_t)BT * QN * 2);
  bf16*  vTb    = (bf16*)alloc((size_t)BT * OUT_N * 2);
  bf16*  attn_b = (bf16*)alloc((size_t)BT * OUT_N * 2);

  // 1. Precision conversion (x) and weight transpose+convert to N-major bf16
  {
    size_t n = (size_t)BT * D_MODEL;
    convert_bf16_kernel<<<dim3((n + 255) / 256), dim3(256), 0, stream>>>(x, xb, n);
  }
  auto tposs = [&](const float* w, bf16* wt, int K, int N) {
    size_t n = (size_t)K * N;
    transpose_bf16_kernel<<<dim3((n + 255) / 256), dim3(256), 0, stream>>>(w, wt, K, N);
  };
  tposs(q_a_w,  wqa_t,  D_MODEL, Q_LORA);
  tposs(q_b_w,  wqb_t,  Q_LORA,  QN);
  tposs(kv_a_w, wkva_t, D_MODEL, KV_LORA + 64);
  tposs(kv_b_w, wkvb_t, KV_LORA, KVN);
  tposs(o_w,    wo_t,   OUT_N,   D_MODEL);

  auto gemm = [&](const bf16* A, const bf16* Bt, float* C, int M, int N, int K) {
    gemm_bf16_nt<<<dim3(N / 64, M / 128), dim3(256), 0, stream>>>(A, Bt, C, M, N, K);
  };

  // 2. q_a = x @ q_a_w ; rmsnorm ; q = qa_n @ q_b_w
  gemm(xb, wqa_t, qa_f, BT, Q_LORA, D_MODEL);
  rmsnorm_bf16_kernel<<<dim3(BT), dim3(256), 0, stream>>>(qa_f, q_a_nw, qa_n,
                                                          Q_LORA, Q_LORA);
  gemm(qa_n, wqb_t, q_f, BT, QN, Q_LORA);

  // 3. kv = x @ kv_a_w ; rmsnorm(first 512) ; kvb = kv_n @ kv_b_w
  gemm(xb, wkva_t, kv_f, BT, KV_LORA + 64, D_MODEL);
  rmsnorm_bf16_kernel<<<dim3(BT), dim3(256), 0, stream>>>(kv_f, kv_nw, kv_n,
                                                          KV_LORA, KV_LORA + 64);
  gemm(kv_n, wkvb_t, kvb_f, BT, KVN, KV_LORA);

  // 4. RoPE + head-major rearrangement
  build_q_kernel<<<dim3(BT), dim3(256), 0, stream>>>(q_f, qsb);
  build_kv_kernel<<<dim3(BT), dim3(256), 0, stream>>>(kvb_f, kv_f, ksb, vTb);

  // 5. Causal flash attention (WMMA for QK^T and PV)
  mla_attn_kernel<<<dim3(SEQ / 64, BATCH * NHEADS), dim3(128), 0, stream>>>(
      qsb, ksb, vTb, attn_b);

  // 6. Output projection
  gemm(attn_b, wo_t, out, BT, D_MODEL, OUT_N);
}
